// LSTMAutoEncoder_33973191311406
// MI455X (gfx1250) — compile-verified, matrix operand order checked by reading, not run on hardware
//
#include <hip/hip_runtime.h>

typedef __attribute__((ext_vector_type(16))) __bf16 v16bf;
typedef __attribute__((ext_vector_type(8)))  float  v8f;

#define T_STEPS 512
#define BATCH   256
#define DIN     32
#define HID     512

#define V8F_ZERO ((v8f){0.f, 0.f, 0.f, 0.f, 0.f, 0.f, 0.f, 0.f})

__device__ __forceinline__ float sigm(float x) {
  return __builtin_amdgcn_rcpf(1.0f + __expf(-x));
}
__device__ __forceinline__ float ftanh(float x) {
  float e = __expf(2.0f * x);
  return 1.0f - 2.0f * __builtin_amdgcn_rcpf(e + 1.0f);
}
__device__ __forceinline__ v8f wmma_bf16(v16bf a, v16bf b, v8f c) {
  return __builtin_amdgcn_wmma_f32_16x16x32_bf16(false, a, false, b, (short)0, c,
                                                 false, false);
}

// Opaque zero offset in an SGPR. The volatile asm executes every timestep, so
// addresses formed with it are not loop-invariant -> LICM cannot hoist the
// weight-fragment loads out of the 512-step loop (which previously caused a
// massive scratch spill). Unlike laundering the *pointer*, adding an opaque
// *offset* preserves address-space provenance, so the loads stay
// global_load_b128 (saddr form) instead of degrading to flat_load_b128.
__device__ __forceinline__ size_t opaque_zero() {
  size_t z = 0;
  asm volatile("" : "+s"(z));
  return z;
}

// ---------------------------------------------------------------------------
// Pre-swizzle fp32 row-major [R,K] (rows = GEMM output columns, B[k][n] =
// src[n][k]) into bf16 WMMA B-fragment blocks of 32 lanes x 16 bf16.
// ---------------------------------------------------------------------------
__global__ void swz_b_kernel(const float* __restrict__ src, __bf16* __restrict__ dst,
                             int R, int K) {
  int idx = blockIdx.x * blockDim.x + threadIdx.x;
  if (idx >= R * K) return;
  int s    = idx & 511;
  int blk  = idx >> 9;
  int KB   = K >> 5;
  int nt   = blk / KB;
  int kb   = blk - nt * KB;
  int lane = s >> 4;
  int j    = s & 15;
  int row  = nt * 16 + (lane & 15);
  int k    = kb * 32 + (j & 7) + ((j & 8) ? 16 : 0) + ((lane & 16) ? 8 : 0);
  dst[idx] = (__bf16)src[row * K + k];
}

// Recurrent GEMM over 16 k-blocks, software-pipelined: the A-fragment and the
// 4 B-fragments of k-block kb+1 are in flight while the 4 WMMAs of kb execute.
__device__ __forceinline__ void recurrent_mm(v8f acc[4], int lane, int w,
                                             const __bf16 (&hb)[16][32][16],
                                             const __bf16* __restrict__ Whh) {
  const __bf16* wp = Whh + opaque_zero() + ((size_t)w << 13) + (lane << 4);
  v16bf a_cur = *(const v16bf*)&hb[0][lane][0];
  v16bf b0 = *(const v16bf*)(wp);
  v16bf b1 = *(const v16bf*)(wp + (1u << 18));  // gate stride 32*8192 elements
  v16bf b2 = *(const v16bf*)(wp + (2u << 18));
  v16bf b3 = *(const v16bf*)(wp + (3u << 18));
#pragma unroll
  for (int kb = 0; kb < 16; ++kb) {
    v16bf a_nxt, n0, n1, n2, n3;
    if (kb < 15) {
      const __bf16* wn = wp + (size_t)(kb + 1) * 512;
      a_nxt = *(const v16bf*)&hb[kb + 1][lane][0];
      n0 = *(const v16bf*)(wn);
      n1 = *(const v16bf*)(wn + (1u << 18));
      n2 = *(const v16bf*)(wn + (2u << 18));
      n3 = *(const v16bf*)(wn + (3u << 18));
    }
    acc[0] = wmma_bf16(a_cur, b0, acc[0]);
    acc[1] = wmma_bf16(a_cur, b1, acc[1]);
    acc[2] = wmma_bf16(a_cur, b2, acc[2]);
    acc[3] = wmma_bf16(a_cur, b3, acc[3]);
    if (kb < 15) { a_cur = a_nxt; b0 = n0; b1 = n1; b2 = n2; b3 = n3; }
  }
}

// Input-term: acc[g] += x[16x32] @ Wih_g (one k-block).
__device__ __forceinline__ void input_mm(v8f acc[4], int lane, int w,
                                         const __bf16 (&xb)[32][16],
                                         const __bf16* __restrict__ Wih) {
  const __bf16* xp = Wih + opaque_zero() + ((size_t)w << 9) + (lane << 4);
  v16bf ax = *(const v16bf*)&xb[lane][0];
  v16bf x0 = *(const v16bf*)(xp);
  v16bf x1 = *(const v16bf*)(xp + (1u << 14));  // gate stride 32*512 elements
  v16bf x2 = *(const v16bf*)(xp + (2u << 14));
  v16bf x3 = *(const v16bf*)(xp + (3u << 14));
  acc[0] = wmma_bf16(ax, x0, acc[0]);
  acc[1] = wmma_bf16(ax, x1, acc[1]);
  acc[2] = wmma_bf16(ax, x2, acc[2]);
  acc[3] = wmma_bf16(ax, x3, acc[3]);
}

// ---------------------------------------------------------------------------
// Persistent LSTM autoencoder: 16 WGs x 16 batch rows, 1024 thr = 32 waves.
// Wave w owns h-columns [16w,16w+16); its c-tile lives in VGPRs for all 1024
// steps; h double-buffered in LDS in WMMA A-fragment layout. Intra-WG
// barriers only. Decoder hides the y=h@Wl^T GEMM (waves 0-1) under the
// all-wave recurrent GEMM.
// ---------------------------------------------------------------------------
__global__ __launch_bounds__(1024, 1) void lstm_ae_kernel(
    const float* __restrict__ X,
    const __bf16* __restrict__ WihE, const __bf16* __restrict__ WhhE, const float* __restrict__ bE,
    const __bf16* __restrict__ WihD, const __bf16* __restrict__ WhhD, const float* __restrict__ bD,
    const __bf16* __restrict__ Wl,   const float* __restrict__ bl,
    float* __restrict__ out) {
  __shared__ __align__(32) __bf16 hbuf[2][16][32][16];
  __shared__ __align__(32) __bf16 xbuf[2][32][16];

  const int tid  = threadIdx.x;
  const int w    = tid >> 5;
  const int lane = tid & 31;
  const int b0   = blockIdx.x << 4;
  const int ncol = (w << 4) + (lane & 15);

  float biasE[4], biasD[4];
#pragma unroll
  for (int g = 0; g < 4; ++g) {
    biasE[g] = bE[g * HID + ncol];
    biasD[g] = bD[g * HID + ncol];
  }

  const int kbh    = ncol >> 5;
  const int kk     = ncol & 31;
  const int jj     = (kk & 7) | ((kk & 16) >> 1);
  const int ltbase = ((kk & 8) ? 16 : 0) + ((lane >> 4) << 3);

  {
    __bf16* hp = &hbuf[0][0][0][0];
    for (int i = tid; i < 16 * 32 * 16; i += 1024) hp[i] = (__bf16)0.0f;
  }
  if (tid < 512) {
    int lt = tid >> 4, j = tid & 15;
    int M  = lt & 15;
    int k  = (j & 7) + ((j & 8) ? 16 : 0) + ((lt & 16) ? 8 : 0);
    xbuf[0][lt][j] = (__bf16)__builtin_nontemporal_load(&X[(size_t)(b0 + M) * DIN + k]);
  }
  v8f c = V8F_ZERO;
  __syncthreads();

  // ---------------- encoder: 1 barrier/step ----------------
  for (int t = 0; t < T_STEPS; ++t) {
    const int cb = t & 1, nb = cb ^ 1;
    v8f acc[4] = {V8F_ZERO, V8F_ZERO, V8F_ZERO, V8F_ZERO};
    recurrent_mm(acc, lane, w, hbuf[cb], WhhE);
    input_mm(acc, lane, w, xbuf[cb], WihE);
#pragma unroll
    for (int v = 0; v < 8; ++v) {
      float iv = sigm(acc[0][v] + biasE[0]);
      float fv = sigm(acc[1][v] + biasE[1]);
      float gv = ftanh(acc[2][v] + biasE[2]);
      float ov = sigm(acc[3][v] + biasE[3]);
      float cv = fv * c[v] + iv * gv;
      c[v] = cv;
      hbuf[nb][kbh][ltbase + v][jj] = (__bf16)(ov * ftanh(cv));
    }
    if (t + 1 < T_STEPS && tid < 512) {
      int lt = tid >> 4, j = tid & 15;
      int M  = lt & 15;
      int k  = (j & 7) + ((j & 8) ? 16 : 0) + ((lt & 16) ? 8 : 0);
      xbuf[nb][lt][j] =
          (__bf16)__builtin_nontemporal_load(&X[(size_t)((t + 1) * BATCH + b0 + M) * DIN + k]);
    }
    __syncthreads();
  }

  // ---------------- decoder: 2 barriers/step, y-GEMM hidden ----------------
  if (tid < 512) xbuf[0][tid >> 4][tid & 15] = (__bf16)0.0f;  // x(0) = 0
  __syncthreads();

  const int d    = (w << 4) + (lane & 15);
  const int M0   = (lane >> 4) << 3;
  const int j2   = (d & 7) | ((d & 16) >> 1);
  const int lt2  = ((d & 8) ? 16 : 0) + M0;
  const float bb = (w < 2) ? bl[d] : 0.f;

  int cur = 0;  // final encoder h is in hbuf[0] (T even)
  for (int t = 0; t < T_STEPS; ++t) {
    const int nb = cur ^ 1;
    // Phase A: waves 0-1 compute y_{t-1} from hbuf[cur] and publish x_t;
    // all waves run the recurrent GEMM on hbuf[cur] concurrently.
    if (w < 2 && t > 0) {
      const __bf16* wlp = Wl + opaque_zero() + ((size_t)(w << 4) << 9) + (lane << 4);
      v8f y = V8F_ZERO;
      v16bf a_cur = *(const v16bf*)&hbuf[cur][0][lane][0];
      v16bf bw = *(const v16bf*)(wlp);
#pragma unroll
      for (int kb = 0; kb < 16; ++kb) {
        v16bf a_nxt, bn;
        if (kb < 15) {
          a_nxt = *(const v16bf*)&hbuf[cur][kb + 1][lane][0];
          bn    = *(const v16bf*)(wlp + (size_t)(kb + 1) * 512);
        }
        y = wmma_bf16(a_cur, bw, y);
        if (kb < 15) { a_cur = a_nxt; bw = bn; }
      }
#pragma unroll
      for (int v = 0; v < 8; ++v) {
        float yv = y[v] + bb;
        __builtin_nontemporal_store(yv, &out[(size_t)((t - 1) * BATCH + b0 + M0 + v) * DIN + d]);
        xbuf[0][lt2 + v][j2] = (__bf16)yv;  // x_t for phase B
      }
    }
    v8f acc[4] = {V8F_ZERO, V8F_ZERO, V8F_ZERO, V8F_ZERO};
    recurrent_mm(acc, lane, w, hbuf[cur], WhhD);
    __syncthreads();  // x_t visible

    // Phase B: x-term (fresh x_t) + cell update
    input_mm(acc, lane, w, xbuf[0], WihD);
#pragma unroll
    for (int v = 0; v < 8; ++v) {
      float iv = sigm(acc[0][v] + biasD[0]);
      float fv = sigm(acc[1][v] + biasD[1]);
      float gv = ftanh(acc[2][v] + biasD[2]);
      float ov = sigm(acc[3][v] + biasD[3]);
      float cv = fv * c[v] + iv * gv;
      c[v] = cv;
      hbuf[nb][kbh][ltbase + v][jj] = (__bf16)(ov * ftanh(cv));
    }
    __syncthreads();  // h_{t+1} visible
    cur = nb;
  }

  // Final output y_{T-1} from the last hidden state.
  if (w < 2) {
    const __bf16* wlp = Wl + ((size_t)(w << 4) << 9) + (lane << 4);
    v8f y = V8F_ZERO;
#pragma unroll
    for (int kb = 0; kb < 16; ++kb) {
      v16bf ah = *(const v16bf*)&hbuf[cur][kb][lane][0];
      v16bf bw = *(const v16bf*)(wlp + (size_t)kb * 512);
      y = wmma_bf16(ah, bw, y);
    }
#pragma unroll
    for (int v = 0; v < 8; ++v) {
      float yv = y[v] + bb;
      __builtin_nontemporal_store(yv, &out[(size_t)((T_STEPS - 1) * BATCH + b0 + M0 + v) * DIN + d]);
    }
  }
}

extern "C" void kernel_launch(void* const* d_in, const int* in_sizes, int n_in,
                              void* d_out, int out_size, void* d_ws, size_t ws_size,
                              hipStream_t stream) {
  const float* X    = (const float*)d_in[0];
  const float* WihE = (const float*)d_in[1];
  const float* WhhE = (const float*)d_in[2];
  const float* bE   = (const float*)d_in[3];
  const float* WihD = (const float*)d_in[4];
  const float* WhhD = (const float*)d_in[5];
  const float* bD   = (const float*)d_in[6];
  const float* Wl   = (const float*)d_in[7];
  const float* bl   = (const float*)d_in[8];
  float* out = (float*)d_out;

  char* ws = (char*)d_ws;
  __bf16* whhE = (__bf16*)(ws);
  __bf16* whhD = (__bf16*)(ws + 2u * 1024u * 1024u);
  __bf16* wihE = (__bf16*)(ws + 4u * 1024u * 1024u);
  __bf16* wihD = (__bf16*)(ws + 4u * 1024u * 1024u + 131072u);
  __bf16* wl   = (__bf16*)(ws + 4u * 1024u * 1024u + 262144u);

  swz_b_kernel<<<(2048 * 512 + 255) / 256, 256, 0, stream>>>(WhhE, whhE, 2048, 512);
  swz_b_kernel<<<(2048 * 512 + 255) / 256, 256, 0, stream>>>(WhhD, whhD, 2048, 512);
  swz_b_kernel<<<(2048 * 32 + 255) / 256, 256, 0, stream>>>(WihE, wihE, 2048, 32);
  swz_b_kernel<<<(2048 * 32 + 255) / 256, 256, 0, stream>>>(WihD, wihD, 2048, 32);
  swz_b_kernel<<<(32 * 512 + 255) / 256, 256, 0, stream>>>(Wl, wl, 32, 512);

  lstm_ae_kernel<<<16, 1024, 0, stream>>>(X, wihE, whhE, bE, wihD, whhD, bD, wl, bl, out);
}
